// EncTransformer_33913061769245
// MI455X (gfx1250) — matrix operand order repaired and checked by reading, not compiled, not run on hardware
//
#include <hip/hip_runtime.h>

typedef __attribute__((ext_vector_type(16))) _Float16 v16h;
typedef __attribute__((ext_vector_type(8)))  _Float16 v8h;
typedef __attribute__((ext_vector_type(8)))  float    v8f;

#define NG   128          // graphs (virtual nodes)
#define NA   64           // atoms per graph
#define DIM  256
#define NH   8
#define DKH  32           // per-head dim
#define FFD  1024
#define NL   4
#define NT   (NG + NG*NA) // 8320 tokens total
#define TQ   80           // padded query rows  (65 -> 80)
#define TKY  96           // padded key rows    (65 -> 96, multiple of 32 for PV)

// ---------------------------------------------------------------- embedding
__global__ __launch_bounds__(256)
void embed_kernel(const float* __restrict__ table, const int* __restrict__ types,
                  float* __restrict__ x)
{
    const int row = blockIdx.x, d = threadIdx.x;
    const int t = (row < NG) ? 0 : types[row - NG];
    x[(size_t)row * DIM + d] = table[(size_t)t * DIM + d];
}

// ---------------------------------------------------------------- layernorm
__global__ __launch_bounds__(256)
void ln_kernel(const float* __restrict__ x, const float* __restrict__ w,
               const float* __restrict__ b, _Float16* __restrict__ out16,
               float* __restrict__ out32)
{
    __shared__ float red[256];
    const int row = blockIdx.x, tid = threadIdx.x;
    const float val = x[(size_t)row * DIM + tid];
    red[tid] = val; __syncthreads();
    for (int s = 128; s > 0; s >>= 1) { if (tid < s) red[tid] += red[tid + s]; __syncthreads(); }
    const float mean = red[0] * (1.0f / DIM);
    __syncthreads();
    const float d = val - mean;
    red[tid] = d * d; __syncthreads();
    for (int s = 128; s > 0; s >>= 1) { if (tid < s) red[tid] += red[tid + s]; __syncthreads(); }
    const float var = red[0] * (1.0f / DIM);
    const float y = d * rsqrtf(var + 1e-5f) * w[tid] + b[tid];
    if (out16) out16[(size_t)row * DIM + tid] = (_Float16)y;
    if (out32) out32[(size_t)row * DIM + tid] = y;
}

// ------------------------------------------- fp32 [K,N] -> f16 Bt[N][K] (transposed)
__global__ __launch_bounds__(256)
void convT_kernel(const float* __restrict__ W, _Float16* __restrict__ Bt, int K, int N)
{
    const int idx = blockIdx.x * blockDim.x + threadIdx.x;
    if (idx >= K * N) return;
    const int n = idx / K, kk = idx % K;
    Bt[idx] = (_Float16)W[(size_t)kk * N + n];
}

// ---------------------------------------------------------------- WMMA GEMM
// C[M,N] = op(A[M,K]_f16 @ Bt[N,K]_f16 + bias) (+ res) ; op = optional ReLU
// one wave per 16x64 output strip: A fragment loaded once per k-step,
// reused across 4 WMMAs (4x less A traffic, AI ~6.4 flop/B vs 4)
__global__ __launch_bounds__(256)
void gemm_wmma(const _Float16* __restrict__ A, const _Float16* __restrict__ Bt,
               const float* __restrict__ bias, const float* __restrict__ res,
               float* __restrict__ outF, _Float16* __restrict__ outH,
               int M, int N, int K, int relu)
{
    const int stripsN = N >> 6;                    // 64-wide strips
    const int strips  = (M >> 4) * stripsN;
    const int wid = (blockIdx.x * blockDim.x + threadIdx.x) >> 5;
    if (wid >= strips) return;                     // wave-uniform
    const int lane = threadIdx.x & 31;
    const int l16 = lane & 15;
    const int hw  = lane >> 4;
    const int tM = wid / stripsN, sN = wid % stripsN;

    const _Float16* arow = A  + (size_t)(tM * 16 + l16) * K;
    const _Float16* brow = Bt + (size_t)(sN * 64 + l16) * K;
    const size_t bstep = (size_t)16 * K;           // 16 Bt rows between sub-tiles

    v8f acc[4];
    #pragma unroll
    for (int j = 0; j < 4; ++j) acc[j] = (v8f){0.f, 0.f, 0.f, 0.f, 0.f, 0.f, 0.f, 0.f};

    for (int k0 = 0; k0 < K; k0 += 32) {
        v8h a0 = *(const v8h*)(arow + k0 + hw * 8);
        v8h a1 = *(const v8h*)(arow + k0 + 16 + hw * 8);
        v16h a;
        #pragma unroll
        for (int i = 0; i < 8; ++i) { a[i] = a0[i]; a[i + 8] = a1[i]; }
        #pragma unroll
        for (int j = 0; j < 4; ++j) {
            v8h b0 = *(const v8h*)(brow + j * bstep + k0 + hw * 8);
            v8h b1 = *(const v8h*)(brow + j * bstep + k0 + 16 + hw * 8);
            v16h b;
            #pragma unroll
            for (int i = 0; i < 8; ++i) { b[i] = b0[i]; b[i + 8] = b1[i]; }
            acc[j] = __builtin_amdgcn_wmma_f32_16x16x32_f16(false, a, false, b, (short)0, acc[j], false, false);
        }
    }
    #pragma unroll
    for (int j = 0; j < 4; ++j) {
        const int col = sN * 64 + j * 16 + l16;
        const float bv = bias ? bias[col] : 0.f;
        #pragma unroll
        for (int i = 0; i < 8; ++i) {
            const int row = tM * 16 + i + hw * 8;
            float v = acc[j][i] + bv;
            if (relu) v = fmaxf(v, 0.f);
            if (res)  v += res[(size_t)row * N + col];
            if (outF) outF[(size_t)row * N + col] = v;
            if (outH) outH[(size_t)row * N + col] = (_Float16)v;
        }
    }
}

// ---------------------------------------------------------------- attention
// one workgroup = one (graph, head); full 65-token self-attention via WMMA
__global__ __launch_bounds__(256)
void attn_wmma(const _Float16* __restrict__ q, const _Float16* __restrict__ k,
               const _Float16* __restrict__ v, _Float16* __restrict__ agg,
               float scale)
{
    __shared__ __align__(16) _Float16 Qs[TQ * DKH];   // 80x32  (5120 B)
    __shared__ __align__(16) _Float16 Ks[TKY * DKH];  // 96x32  (6144 B)
    __shared__ __align__(16) _Float16 Vt[DKH * TKY];  // 32x96  (6144 B)  V^T
    __shared__ __align__(16) float    S[TQ * TKY];    // 80x96  (30720 B)
    __shared__ __align__(16) _Float16 P[TQ * TKY];    // 80x96  (15360 B)

    const int g = blockIdx.x, h = blockIdx.y;
    const int tid = threadIdx.x;

    // stage Q (80x32) and K (96x32), zero padding beyond 65 real tokens
    for (int idx = tid; idx < TKY * 4; idx += 256) {
        const int t = idx >> 2, c = idx & 3;
        const int node = (t == 0) ? g : ((t <= NA) ? NG + g * NA + (t - 1) : -1);
        uint4 kv = {0, 0, 0, 0}, qv = {0, 0, 0, 0};
        if (node >= 0) {
            kv = *(const uint4*)(k + (size_t)node * DIM + h * DKH + c * 8);
            if (t < TQ) qv = *(const uint4*)(q + (size_t)node * DIM + h * DKH + c * 8);
        }
        *(uint4*)(Ks + t * DKH + c * 8) = kv;
        if (t < TQ) *(uint4*)(Qs + t * DKH + c * 8) = qv;
    }
    // stage V transposed: Vt[dk][t]
    for (int idx = tid; idx < TKY * DKH; idx += 256) {
        const int t = idx >> 5, dk = idx & 31;
        const int node = (t == 0) ? g : ((t <= NA) ? NG + g * NA + (t - 1) : -1);
        _Float16 val = (_Float16)0.f;
        if (node >= 0) val = v[(size_t)node * DIM + h * DKH + dk];
        Vt[dk * TKY + t] = val;
    }
    __syncthreads();

    const int w = tid >> 5, lane = tid & 31, l16 = lane & 15, hw = lane >> 4;

    // S = scale * Q @ K^T : 5x6 tiles of 16x16, K-dim = 32 (one WMMA each)
    for (int tile = w; tile < 30; tile += 8) {
        const int tm = tile / 6, tn = tile % 6;
        v8h a0 = *(const v8h*)(Qs + (tm * 16 + l16) * DKH + hw * 8);
        v8h a1 = *(const v8h*)(Qs + (tm * 16 + l16) * DKH + 16 + hw * 8);
        v8h b0 = *(const v8h*)(Ks + (tn * 16 + l16) * DKH + hw * 8);
        v8h b1 = *(const v8h*)(Ks + (tn * 16 + l16) * DKH + 16 + hw * 8);
        v16h a, b;
        #pragma unroll
        for (int i = 0; i < 8; ++i) { a[i] = a0[i]; a[i + 8] = a1[i]; b[i] = b0[i]; b[i + 8] = b1[i]; }
        v8f acc = {0.f, 0.f, 0.f, 0.f, 0.f, 0.f, 0.f, 0.f};
        acc = __builtin_amdgcn_wmma_f32_16x16x32_f16(false, a, false, b, (short)0, acc, false, false);
        const int col = tn * 16 + l16;
        #pragma unroll
        for (int i = 0; i < 8; ++i) {
            const int row = tm * 16 + i + hw * 8;
            float s = acc[i] * scale;
            if (col > NA) s = -3.0e38f;           // mask padded keys
            S[row * TKY + col] = s;
        }
    }
    __syncthreads();

    // per-row softmax over 96 keys (padded keys contribute exp -> 0)
    if (tid < TQ) {
        float m = -3.0e38f;
        for (int c = 0; c < TKY; ++c) m = fmaxf(m, S[tid * TKY + c]);
        float sum = 0.f;
        for (int c = 0; c < TKY; ++c) { float e = expf(S[tid * TKY + c] - m); S[tid * TKY + c] = e; sum += e; }
        const float inv = 1.f / sum;
        for (int c = 0; c < TKY; ++c) P[tid * TKY + c] = (_Float16)(S[tid * TKY + c] * inv);
    }
    __syncthreads();

    // agg = P @ V : 5x2 tiles, 3 K-steps of 32
    for (int tile = w; tile < 10; tile += 8) {
        const int tm = tile >> 1, tn = tile & 1;
        v8f acc = {0.f, 0.f, 0.f, 0.f, 0.f, 0.f, 0.f, 0.f};
        #pragma unroll
        for (int kc = 0; kc < 3; ++kc) {
            v8h a0 = *(const v8h*)(P + (tm * 16 + l16) * TKY + kc * 32 + hw * 8);
            v8h a1 = *(const v8h*)(P + (tm * 16 + l16) * TKY + kc * 32 + 16 + hw * 8);
            v8h b0 = *(const v8h*)(Vt + (tn * 16 + l16) * TKY + kc * 32 + hw * 8);
            v8h b1 = *(const v8h*)(Vt + (tn * 16 + l16) * TKY + kc * 32 + 16 + hw * 8);
            v16h a, b;
            #pragma unroll
            for (int i = 0; i < 8; ++i) { a[i] = a0[i]; a[i + 8] = a1[i]; b[i] = b0[i]; b[i + 8] = b1[i]; }
            acc = __builtin_amdgcn_wmma_f32_16x16x32_f16(false, a, false, b, (short)0, acc, false, false);
        }
        const int dk = tn * 16 + l16;
        #pragma unroll
        for (int i = 0; i < 8; ++i) {
            const int t = tm * 16 + i + hw * 8;
            if (t <= NA) {
                const int node = (t == 0) ? g : NG + g * NA + (t - 1);
                agg[(size_t)node * DIM + h * DKH + dk] = (_Float16)acc[i];
            }
        }
    }
}

// ---------------------------------------------------------------- launcher
extern "C" void kernel_launch(void* const* d_in, const int* in_sizes, int n_in,
                              void* d_out, int out_size, void* d_ws, size_t ws_size,
                              hipStream_t stream)
{
    const float* embed = (const float*)d_in[0];
    const float* Wq = (const float*)d_in[1];  const float* bq = (const float*)d_in[2];
    const float* Wk = (const float*)d_in[3];  const float* bk = (const float*)d_in[4];
    const float* Wv = (const float*)d_in[5];  const float* bv = (const float*)d_in[6];
    const float* Wo = (const float*)d_in[7];  const float* bo = (const float*)d_in[8];
    const float* ln1w = (const float*)d_in[9];  const float* ln1b = (const float*)d_in[10];
    const float* ln2w = (const float*)d_in[11]; const float* ln2b = (const float*)d_in[12];
    const float* f1w = (const float*)d_in[13]; const float* f1b = (const float*)d_in[14];
    const float* f2w = (const float*)d_in[15]; const float* f2b = (const float*)d_in[16];
    const float* lnfw = (const float*)d_in[17]; const float* lnfb = (const float*)d_in[18];
    const int* atom_types = (const int*)d_in[19];
    float* out = (float*)d_out;

    char* ws = (char*)d_ws;
    size_t off = 0;
    auto alloc = [&](size_t bytes) {
        char* p = ws + off;
        off = (off + bytes + 255) & ~(size_t)255;
        return p;
    };
    float*    x     = (float*)   alloc((size_t)NT * DIM * 4);
    _Float16* h16   = (_Float16*)alloc((size_t)NT * DIM * 2);
    _Float16* q16   = (_Float16*)alloc((size_t)NT * DIM * 2);
    _Float16* k16   = (_Float16*)alloc((size_t)NT * DIM * 2);
    _Float16* v16   = (_Float16*)alloc((size_t)NT * DIM * 2);
    _Float16* agg16 = (_Float16*)alloc((size_t)NT * DIM * 2);
    _Float16* ff16  = (_Float16*)alloc((size_t)NT * FFD * 2);
    const size_t LW = 4 * (size_t)DIM * DIM + 2 * (size_t)DIM * FFD;  // 786432 halves / layer
    _Float16* W16   = (_Float16*)alloc((size_t)NL * LW * 2);

    // one-time weight conversion (fp32 [K,N] -> f16 Bt[N,K])
    const int blkDD = (DIM * DIM + 255) / 256;
    const int blkDF = (DIM * FFD + 255) / 256;
    for (int l = 0; l < NL; ++l) {
        _Float16* base = W16 + (size_t)l * LW;
        convT_kernel<<<blkDD, 256, 0, stream>>>(Wq + (size_t)l * DIM * DIM, base + 0,          DIM, DIM);
        convT_kernel<<<blkDD, 256, 0, stream>>>(Wk + (size_t)l * DIM * DIM, base + 65536,      DIM, DIM);
        convT_kernel<<<blkDD, 256, 0, stream>>>(Wv + (size_t)l * DIM * DIM, base + 131072,     DIM, DIM);
        convT_kernel<<<blkDD, 256, 0, stream>>>(Wo + (size_t)l * DIM * DIM, base + 196608,     DIM, DIM);
        convT_kernel<<<blkDF, 256, 0, stream>>>(f1w + (size_t)l * DIM * FFD, base + 262144,    DIM, FFD);
        convT_kernel<<<blkDF, 256, 0, stream>>>(f2w + (size_t)l * FFD * DIM, base + 524288,    FFD, DIM);
    }

    embed_kernel<<<NT, 256, 0, stream>>>(embed, atom_types, x);

    const float scale = 0.17677669529663687f;  // 1/sqrt(32)
    const int blkD = (((NT / 16) * (DIM / 64)) + 7) / 8;   // 2080 strips -> 260 blocks
    const int blkF = (((NT / 16) * (FFD / 64)) + 7) / 8;   // 8320 strips -> 1040 blocks

    for (int l = 0; l < NL; ++l) {
        _Float16* base = W16 + (size_t)l * LW;
        ln_kernel<<<NT, 256, 0, stream>>>(x, ln1w + l * DIM, ln1b + l * DIM, h16, nullptr);
        gemm_wmma<<<blkD, 256, 0, stream>>>(h16, base + 0,      bq + l * DIM, nullptr, nullptr, q16, NT, DIM, DIM, 0);
        gemm_wmma<<<blkD, 256, 0, stream>>>(h16, base + 65536,  bk + l * DIM, nullptr, nullptr, k16, NT, DIM, DIM, 0);
        gemm_wmma<<<blkD, 256, 0, stream>>>(h16, base + 131072, bv + l * DIM, nullptr, nullptr, v16, NT, DIM, DIM, 0);
        attn_wmma<<<dim3(NG, NH), 256, 0, stream>>>(q16, k16, v16, agg16, scale);
        gemm_wmma<<<blkD, 256, 0, stream>>>(agg16, base + 196608, bo + l * DIM, x, x, nullptr, NT, DIM, DIM, 0);
        ln_kernel<<<NT, 256, 0, stream>>>(x, ln2w + l * DIM, ln2b + l * DIM, h16, nullptr);
        gemm_wmma<<<blkF, 256, 0, stream>>>(h16,  base + 262144, f1b + l * FFD, nullptr, nullptr, ff16, NT, FFD, DIM, 1);
        gemm_wmma<<<blkD, 256, 0, stream>>>(ff16, base + 524288, f2b + l * DIM, x, x, nullptr, NT, DIM, FFD, 1);
    }

    // final LN: rows 0..127 are exactly the virtual nodes -> d_out [128, 256] fp32
    ln_kernel<<<NG, 256, 0, stream>>>(x, lnfw, lnfb, nullptr, out);
}
